// GRU_1803886265843
// MI455X (gfx1250) — compile-verified, matrix operand order checked by reading, not run on hardware
//
#include <hip/hip_runtime.h>
#include <hip/hip_bf16.h>

// ---------------- types ----------------
typedef __attribute__((ext_vector_type(16))) __bf16 v16bf;
typedef __attribute__((ext_vector_type(8)))  float  v8f;
typedef __attribute__((ext_vector_type(8)))  unsigned int v8u;
typedef __attribute__((ext_vector_type(4)))  unsigned int v4u;
typedef __attribute__((ext_vector_type(8)))  int v8i;
typedef __attribute__((ext_vector_type(4)))  int v4i;

enum : int { cT = 512, cB = 64, cI = 256, cH = 1024, cO = 256, cL = 3 };

// ---------------- helpers ----------------
__device__ __forceinline__ unsigned short f2bf(float f) {
    unsigned int u = __builtin_bit_cast(unsigned int, f);
    unsigned int r = u + 0x7FFFu + ((u >> 16) & 1u);
    return (unsigned short)(r >> 16);
}

__device__ __forceinline__ float sigmoidf_(float x) {
    return 1.0f / (1.0f + __expf(-x));
}

// Load one 16-bit WMMA operand fragment (16 bf16 per lane = two 16B chunks).
// Works for global or LDS pointers (addrspace inferred -> global_load_b128 / ds_load_b128).
__device__ __forceinline__ v16bf load_frag(const unsigned short* p) {
    uint4 lo = *reinterpret_cast<const uint4*>(p);
    uint4 hi = *reinterpret_cast<const uint4*>(p + 16);
    v8u u = { lo.x, lo.y, lo.z, lo.w, hi.x, hi.y, hi.z, hi.w };
    return __builtin_bit_cast(v16bf, u);
}

__device__ __forceinline__ v8f wmma_bf16(v8f acc, v16bf a, v16bf b) {
    // (neg_a, A, neg_b, B, c_mod, C, reuse_a, reuse_b)
    return __builtin_amdgcn_wmma_f32_16x16x32_bf16(false, a, false, b, (short)0, acc, false, false);
}

// TDM: DMA the full hidden state [64 rows x 1024 cols] of bf16 (128KB) into LDS.
// D# per CDNA5 ISA:
//  group0: [1:0]=count=1, [63:32]=lds_addr, [120:64]=global_addr, [127:126]=type=2
//  group1: data_size=1(2B)<<16 ; tensor_dim0=1024 ; tensor_dim1=64 ;
//          tile_dim0=1024 ; tile_dim1=64 ; tensor_dim0_stride=1024
__device__ __forceinline__ void tdm_load_h_to_lds(const unsigned short* gsrc, unsigned lds_off) {
    unsigned long long ga = (unsigned long long)gsrc;
    v4u g0 = { 1u,
               lds_off,
               (unsigned)(ga & 0xFFFFFFFFull),
               (unsigned)(((ga >> 32) & 0x01FFFFFFull) | 0x80000000u) };
    v8i g1 = { (int)0x00010000,      // wg_mask=0, data_size=1 (2 bytes)
               (int)0x04000000,      // tensor_dim0[15:0]=1024 in bits[31:16]
               (int)0x00400000,      // tensor_dim0[31:16]=0 | tensor_dim1[15:0]=64 <<16
               (int)0x04000000,      // tensor_dim1[31:16]=0 | tile_dim0=1024 <<16
               64,                   // tile_dim1=64 | tile_dim2=0
               1024,                 // tensor_dim0_stride[31:0]=1024
               0, 0 };
    v4i z4 = { 0, 0, 0, 0 };
    v8i z8 = { 0, 0, 0, 0, 0, 0, 0, 0 };
    // clang-23 / therock arity: (g0, g1, g2, g3, extra, cpol)
    __builtin_amdgcn_tensor_load_to_lds(g0, g1, z4, z4, z8, 0);
}

// Grid-wide sense/epoch barrier in global memory (bar[0]=count, bar[1]=gen).
__device__ __forceinline__ void grid_barrier(unsigned* bar, unsigned nwg) {
    __syncthreads();
    if (threadIdx.x == 0) {
        __threadfence();
        unsigned gen = __hip_atomic_load(&bar[1], __ATOMIC_RELAXED, __HIP_MEMORY_SCOPE_AGENT);
        unsigned prev = __hip_atomic_fetch_add(&bar[0], 1u, __ATOMIC_ACQ_REL, __HIP_MEMORY_SCOPE_AGENT);
        if (prev + 1u == nwg) {
            __hip_atomic_store(&bar[0], 0u, __ATOMIC_RELAXED, __HIP_MEMORY_SCOPE_AGENT);
            __hip_atomic_fetch_add(&bar[1], 1u, __ATOMIC_RELEASE, __HIP_MEMORY_SCOPE_AGENT);
        } else {
            while (__hip_atomic_load(&bar[1], __ATOMIC_ACQUIRE, __HIP_MEMORY_SCOPE_AGENT) == gen) {
                __builtin_amdgcn_s_sleep(1);
            }
        }
        __threadfence();
    }
    __syncthreads();
}

// ---------------- small utility kernels ----------------
__global__ void zerobar_kernel(unsigned* bar) { bar[0] = 0u; bar[1] = 0u; }

__global__ void cvt_kernel(const float* __restrict__ in, unsigned short* __restrict__ out, int n) {
    int i = blockIdx.x * blockDim.x + threadIdx.x;
    if (i < n) out[i] = f2bf(in[i]);
}

__global__ void hinit_kernel(const float* __restrict__ h0l, float* __restrict__ hf,
                             unsigned short* __restrict__ hb, int n) {
    int i = blockIdx.x * blockDim.x + threadIdx.x;
    if (i < n) { float v = h0l[i]; hf[i] = v; hb[i] = f2bf(v); }
}

// ---------------- persistent GRU layer kernel ----------------
// Grid: 16 workgroups x 256 threads. WG g owns hidden units [64g, 64g+64).
// Wave tiling 2(M)x4(N): mg = wave&1 covers batch rows [mg*32, mg*32+32) as two
// 16-row M tiles; ng = wave>>1 covers hidden cols [nb16, nb16+16) for all 3 gates.
// -> B (weight) fragments are shared by only 2 waves (2x duplication, was 4x);
//    A fragments are read from LDS (TDM-staged h), where duplication is cheap.
// Hidden state is double-buffered (h_buf0/h_buf1); each step the current buffer is
// TDM-DMA'd into LDS (128KB). `inplace`!=0 means xin aliases seq_out (fallback):
// adds a mid-step grid barrier.
__global__ __launch_bounds__(256) void gru_layer_kernel(
    const unsigned short* __restrict__ xin, int Iin,
    const unsigned short* __restrict__ w_ih,   // [3H, Iin] bf16
    const unsigned short* __restrict__ w_hh,   // [3H, H]   bf16
    const float* __restrict__ b_ih, const float* __restrict__ b_hh,
    unsigned short* h_buf0, unsigned short* h_buf1,  // [B,H] bf16 ping-pong (t=0 reads buf0)
    float* __restrict__ h_f32,                 // [B, H] f32  (lives in d_out h_final slot)
    unsigned short* __restrict__ seq_out,      // [T, B, H] bf16 (layer output)
    unsigned* __restrict__ bar, int inplace)
{
    __shared__ unsigned short sh_h[cB * cH];   // 128 KB staging of h_prev

    const int lane  = threadIdx.x & 31;
    const int wave  = threadIdx.x >> 5;
    const int mg    = wave & 1;                           // batch half: rows [mg*32, mg*32+32)
    const int ng    = wave >> 1;                          // 16-col group within WG's 64 cols
    const int nb16  = blockIdx.x * 64 + ng * 16;          // hidden-col base (16 cols)
    const int khalf = lane >> 4;
    const int coll  = lane & 15;
    const int ra    = mg * 32 + coll;                     // A-fragment row, M tile 0
    const int rb    = ra + 16;                            // A-fragment row, M tile 1
    const unsigned nwg = gridDim.x;
    const unsigned lds_base = (unsigned)(unsigned long long)(void*)sh_h; // low 32 bits = LDS offset

    unsigned short* hc = h_buf0;   // current (read)
    unsigned short* hn = h_buf1;   // next (write)

    // Per-lane gate biases (time-invariant); one 16-col tile per wave.
    const int col = nb16 + coll;
    const float br  = b_ih[col]            + b_hh[col];
    const float bz  = b_ih[cH + col]       + b_hh[cH + col];
    const float bnx = b_ih[2 * cH + col];
    const float bnh = b_hh[2 * cH + col];

    const v8f zf = {0.f, 0.f, 0.f, 0.f, 0.f, 0.f, 0.f, 0.f};
    const size_t xstride = (size_t)Iin * 2;  // bytes per batch row of x
    const size_t wrow = (size_t)col;         // weight row for this lane's column

    for (int t = 0; t < cT; ++t) {
        // ---- stage h_prev into LDS via Tensor Data Mover (wave 0 issues) ----
        if (wave == 0) {
            tdm_load_h_to_lds(hc, lds_base);
            __builtin_amdgcn_s_wait_tensorcnt(0);
        }
        // prefetch next step's input slab (fully available ahead of time)
        if (t + 1 < cT) {
            const char* nx = (const char*)(xin + (size_t)(t + 1) * cB * Iin);
            __builtin_prefetch(nx + (size_t)threadIdx.x * ((cB * xstride) / 256), 0, 1);
        }
        __syncthreads();

        v8f accR[2], accZ[2], accNH[2], accNX[2];   // index = M tile (0/1)
#pragma unroll
        for (int mt = 0; mt < 2; ++mt) { accR[mt] = zf; accZ[mt] = zf; accNH[mt] = zf; accNX[mt] = zf; }

        // ---- hidden-side projection: h_prev @ w_hh^T (gates r, z, n_h); A from LDS ----
        for (int kb = 0; kb < cH; kb += 32) {
            v16bf a0 = load_frag(sh_h + (size_t)ra * cH + kb + khalf * 8);
            v16bf a1 = load_frag(sh_h + (size_t)rb * cH + kb + khalf * 8);
            const unsigned short* wb = w_hh + wrow * cH + kb + khalf * 8;
            v16bf bR = load_frag(wb);
            v16bf bZ = load_frag(wb + (size_t)cH * cH);
            v16bf bN = load_frag(wb + (size_t)2 * cH * cH);
            accR[0]  = wmma_bf16(accR[0],  a0, bR);
            accR[1]  = wmma_bf16(accR[1],  a1, bR);
            accZ[0]  = wmma_bf16(accZ[0],  a0, bZ);
            accZ[1]  = wmma_bf16(accZ[1],  a1, bZ);
            accNH[0] = wmma_bf16(accNH[0], a0, bN);
            accNH[1] = wmma_bf16(accNH[1], a1, bN);
        }

        // ---- input-side projection: x_t @ w_ih^T (r, z accumulate; n_x separate) ----
        const unsigned short* xr = xin + (size_t)t * cB * Iin;
        for (int kb = 0; kb < Iin; kb += 32) {
            v16bf a0 = load_frag(xr + (size_t)ra * Iin + kb + khalf * 8);
            v16bf a1 = load_frag(xr + (size_t)rb * Iin + kb + khalf * 8);
            const unsigned short* wb = w_ih + wrow * Iin + kb + khalf * 8;
            v16bf bR = load_frag(wb);
            v16bf bZ = load_frag(wb + (size_t)cH * Iin);
            v16bf bN = load_frag(wb + (size_t)2 * cH * Iin);
            accR[0]  = wmma_bf16(accR[0],  a0, bR);
            accR[1]  = wmma_bf16(accR[1],  a1, bR);
            accZ[0]  = wmma_bf16(accZ[0],  a0, bZ);
            accZ[1]  = wmma_bf16(accZ[1],  a1, bZ);
            accNX[0] = wmma_bf16(accNX[0], a0, bN);
            accNX[1] = wmma_bf16(accNX[1], a1, bN);
        }

        // In-place fallback only: everyone must finish reading seq row t before writes.
        if (inplace) grid_barrier(bar, nwg);

        // ---- gates + state update (each output element owned by exactly one lane) ----
#pragma unroll
        for (int mt = 0; mt < 2; ++mt) {
#pragma unroll
            for (int j = 0; j < 8; ++j) {
                int brow = mg * 32 + mt * 16 + khalf * 8 + j;   // D layout: M = j + 8*(lane>>4)
                float r  = sigmoidf_(accR[mt][j] + br);
                float zg = sigmoidf_(accZ[mt][j] + bz);
                float n  = tanhf(accNX[mt][j] + bnx + r * (accNH[mt][j] + bnh));
                size_t hidx = (size_t)brow * cH + col;
                float hp = h_f32[hidx];
                float hn_v = (1.0f - zg) * n + zg * hp;
                h_f32[hidx] = hn_v;
                unsigned short hb = f2bf(hn_v);
                hn[hidx] = hb;
                seq_out[(size_t)(t * cB + brow) * cH + col] = hb;
            }
        }

        // Writes visible grid-wide before next step's TDM read of hn.
        grid_barrier(bar, nwg);

        unsigned short* tmp = hc; hc = hn; hn = tmp;
    }
}

// ---------------- generic bf16 GEMM + bias: out[M,N] = A[M,K] @ W[N,K]^T + bias ----------------
__global__ __launch_bounds__(256) void gemm_bias_kernel(
    const unsigned short* __restrict__ A, const unsigned short* __restrict__ W,
    const float* __restrict__ bias, float* __restrict__ out,
    int M, int N, int K)
{
    const int lane  = threadIdx.x & 31;
    const int wave  = threadIdx.x >> 5;
    const int mrow  = blockIdx.x * 64 + (wave & 3) * 16;
    const int nbase = blockIdx.y * 64 + (wave >> 2) * 32;
    const int khalf = lane >> 4;
    const int coll  = lane & 15;
    const int ar    = mrow + coll;

    const v8f zf = {0.f, 0.f, 0.f, 0.f, 0.f, 0.f, 0.f, 0.f};
    v8f acc[2] = { zf, zf };

    for (int kb = 0; kb < K; kb += 32) {
        v16bf a = load_frag(A + (size_t)ar * K + kb + khalf * 8);
#pragma unroll
        for (int ct = 0; ct < 2; ++ct) {
            const unsigned short* wb = W + (size_t)(nbase + ct * 16 + coll) * K + kb + khalf * 8;
            acc[ct] = wmma_bf16(acc[ct], a, load_frag(wb));
        }
    }
#pragma unroll
    for (int ct = 0; ct < 2; ++ct) {
        int col = nbase + ct * 16 + coll;
        float b = bias[col];
#pragma unroll
        for (int j = 0; j < 8; ++j) {
            int row = mrow + khalf * 8 + j;
            out[(size_t)row * N + col] = acc[ct][j] + b;
        }
    }
}

// ---------------- host ----------------
extern "C" void kernel_launch(void* const* d_in, const int* in_sizes, int n_in,
                              void* d_out, int out_size, void* d_ws, size_t ws_size,
                              hipStream_t stream) {
    (void)in_sizes; (void)n_in; (void)out_size;

    const float* x    = (const float*)d_in[0];   // [T,B,I]
    const float* h0   = (const float*)d_in[1];   // [L,B,H]
    const float* fc_w = (const float*)d_in[2];   // [O,H]
    const float* fc_b = (const float*)d_in[3];   // [O]
    const float* w_ih[cL]; const float* w_hh[cL];
    const float* b_ih[cL]; const float* b_hh[cL];
    for (int l = 0; l < cL; ++l) {
        w_ih[l] = (const float*)d_in[4 + 4 * l + 0];
        w_hh[l] = (const float*)d_in[4 + 4 * l + 1];
        b_ih[l] = (const float*)d_in[4 + 4 * l + 2];
        b_hh[l] = (const float*)d_in[4 + 4 * l + 3];
    }

    // workspace carve-up (256B aligned)
    char* ws = (char*)d_ws;
    size_t off = 0;
    auto take = [&](size_t bytes) -> void* {
        void* p = ws + off;
        off += (bytes + 255) & ~(size_t)255;
        return p;
    };
    unsigned* bar            = (unsigned*)take(256);
    unsigned short* xbf      = (unsigned short*)take((size_t)cT * cB * cI * 2);
    unsigned short* wihbf[cL];
    unsigned short* whhbf[cL];
    for (int l = 0; l < cL; ++l) {
        int iin = (l == 0) ? cI : cH;
        wihbf[l] = (unsigned short*)take((size_t)3 * cH * iin * 2);
        whhbf[l] = (unsigned short*)take((size_t)3 * cH * cH * 2);
    }
    unsigned short* fcwbf    = (unsigned short*)take((size_t)cO * cH * 2);
    unsigned short* hbf0     = (unsigned short*)take((size_t)cB * cH * 2);
    unsigned short* hbf1     = (unsigned short*)take((size_t)cB * cH * 2);
    const size_t seq_bytes   = (size_t)cT * cB * cH * 2;
    unsigned short* seqA     = (unsigned short*)take(seq_bytes);

    // If workspace permits, double-buffer the sequence -> only ONE grid barrier per step.
    bool pp = (off + seq_bytes) <= ws_size;
    unsigned short* seqB = pp ? (unsigned short*)take(seq_bytes) : seqA;

    float* y_out   = (float*)d_out;                         // [T,B,O]
    float* h_final = (float*)d_out + (size_t)cT * cB * cO;  // [L,B,H]

    // reset barrier state every call (graph-replay safe)
    zerobar_kernel<<<1, 1, 0, stream>>>(bar);

    // bf16 conversions
    auto cvt = [&](const float* in, unsigned short* out, int n) {
        cvt_kernel<<<(n + 255) / 256, 256, 0, stream>>>(in, out, n);
    };
    cvt(x, xbf, cT * cB * cI);
    for (int l = 0; l < cL; ++l) {
        int iin = (l == 0) ? cI : cH;
        cvt(w_ih[l], wihbf[l], 3 * cH * iin);
        cvt(w_hh[l], whhbf[l], 3 * cH * cH);
    }
    cvt(fc_w, fcwbf, cO * cH);

    // GRU layers
    const unsigned short* lin[cL];
    unsigned short* lout[cL];
    int linplace[cL];
    if (pp) {
        lin[0] = xbf;  lout[0] = seqA; linplace[0] = 0;
        lin[1] = seqA; lout[1] = seqB; linplace[1] = 0;
        lin[2] = seqB; lout[2] = seqA; linplace[2] = 0;
    } else {
        lin[0] = xbf;  lout[0] = seqA; linplace[0] = 0;
        lin[1] = seqA; lout[1] = seqA; linplace[1] = 1;
        lin[2] = seqA; lout[2] = seqA; linplace[2] = 1;
    }

    for (int l = 0; l < cL; ++l) {
        int iin = (l == 0) ? cI : cH;
        float* hf32 = h_final + (size_t)l * cB * cH;
        int n = cB * cH;
        hinit_kernel<<<(n + 255) / 256, 256, 0, stream>>>(h0 + (size_t)l * cB * cH, hf32, hbf0, n);
        gru_layer_kernel<<<16, 256, 0, stream>>>(lin[l], iin, wihbf[l], whhbf[l],
                                                 b_ih[l], b_hh[l], hbf0, hbf1,
                                                 hf32, lout[l], bar, linplace[l]);
    }

    // final FC: [T*B, H] x [O, H]^T + fc_b ; layer-2 output is seqA in both modes
    dim3 g((cT * cB) / 64, cO / 64);
    gemm_bias_kernel<<<g, 256, 0, stream>>>(seqA, fcwbf, fc_b, y_out, cT * cB, cO, cH);
}
